// AbsorberPathAggregator_77627238908546
// MI455X (gfx1250) — matrix operand order repaired and checked by reading, not compiled, not run on hardware
//
#include <hip/hip_runtime.h>
#include <hip/hip_bf16.h>
#include <math.h>

typedef __attribute__((ext_vector_type(16))) _Float16 v16h;
typedef __attribute__((ext_vector_type(8)))  _Float16 v8h;
typedef __attribute__((ext_vector_type(8)))  float    v8f;

#define DEV static __device__ __forceinline__

// ---- problem constants (from reference) ----
#define BDIM   32
#define NNODE  64
#define DDIM   128
#define NE_    100
#define PMAX_  256
#define SCAT_  32
#define GH_    128
#define PEH_   64
#define OUTD_  64
#define CUTF   6.0f
#define NPAIR  2016
#define PI_F   3.14159265358979f
#define ECHUNK 20      // NE split into 5 chunks of 20 for kernel 3

// ---- workspace layout ----
// f16 region (offsets in halves)
constexpr size_t H_PE1   = 0;                       // 2*4 frags * 512
constexpr size_t H_PE2   = H_PE1 + 2*4*512;         // 4096
constexpr size_t H_PE3   = H_PE2 + 2*4*512;         // 8192
constexpr size_t H_GM1   = H_PE3 + 2*2*512;         // 10240
constexpr size_t H_GM2   = H_GM1 + 12*8*512;        // 59392
constexpr size_t H_GM3   = H_GM2 + 4*8*512;         // 75776
constexpr size_t H_EF    = H_GM3 + 4*2*512;         // 79872
constexpr size_t H_EPAIR = H_EF + NE_*16;           // 81472
constexpr size_t H_TOTAL = H_EPAIR + (size_t)BDIM*PMAX_*32;   // 343616 halves
constexpr size_t FBASE   = ((H_TOTAL*2 + 255)/256)*256;       // byte offset of float region
// float region (offsets in floats)
constexpr size_t F_W     = 0;                       // [B*PMAX]
constexpr size_t F_WSUM  = F_W + BDIM*PMAX_;        // [B]
constexpr size_t F_GG    = F_WSUM + BDIM;           // w-scaled g_geom [B*PMAX*32]
constexpr size_t F_AGG   = F_GG + (size_t)BDIM*PMAX_*SCAT_;
constexpr size_t F_TOTAL = F_AGG + (size_t)BDIM*NE_*SCAT_;
constexpr size_t IBASE   = FBASE + ((F_TOTAL*4 + 255)/256)*256; // int region (jidx,kidx)

// ---- small helpers ----
DEV float siluf(float x) { return x / (1.0f + __expf(-x)); }
DEV float cutoff_fn(float r) {
  return (r < CUTF) ? 0.5f * (__cosf(PI_F * r * (1.0f / CUTF)) + 1.0f) : 0.0f;
}
DEV v8f vzero() { v8f z = {0.f,0.f,0.f,0.f,0.f,0.f,0.f,0.f}; return z; }

DEV v8f wmma32(v16h a, v16h b, v8f c) {
  // v_wmma_f32_16x16x32_f16, D = A*B + C, f32 accumulate
  return __builtin_amdgcn_wmma_f32_16x16x32_f16(false, a, false, b, (short)0, c,
                                                false, false);
}

// A fragment: 16xK row-major tile in LDS. Per ISA A-layout, lane L (row L&15,
// hi=(L>=16)*8) needs K = kt*32 + {hi..hi+7, 16+hi..16+hi+7}: two 16B chunks.
DEV v16h load_a_frag(const _Float16* base, int lane, int stride, int kt) {
  const int row = lane & 15;
  const int hi  = (lane >> 4) << 3;
  const _Float16* p = base + row * stride + kt * 32 + hi;
  v8h lo = *(const v8h*)p;
  v8h hh = *(const v8h*)(p + 16);
  v16h r;
#pragma unroll
  for (int i = 0; i < 8; ++i) { r[i] = lo[i]; r[i + 8] = hh[i]; }
  return r;
}

// B fragment: pre-packed lane-major, 32B contiguous per lane per fragment.
DEV v16h load_b_frag(const _Float16* pack, int frag, int lane) {
  return *(const v16h*)(pack + (size_t)frag * 512 + lane * 16);
}

// D (v8f) -> bias + silu -> f16 tile in LDS. D element r = row (r + 8*hiHalf),
// col = nt*16 + (lane&15).
template <int NT>
DEV void bias_silu_store(const v8f* acc, _Float16* Ht, int stride,
                         const float* bias, int lane) {
  const int cl = lane & 15, rb = (lane >> 4) << 3;
#pragma unroll
  for (int nt = 0; nt < NT; ++nt) {
    const int col = nt * 16 + cl;
    const float bv = bias[col];
#pragma unroll
    for (int r = 0; r < 8; ++r) {
      float x = acc[nt][r] + bv;
      Ht[(rb + r) * stride + col] = (_Float16)siluf(x);
    }
  }
}

// ================= kernel 0a: pack a weight matrix into WMMA B fragments ====
// B layout: element i of lane L = W[kt*32 + (L>=16?8:0) + (i&7) + (i>=8?16:0)][nt*16 + (L&15)],
// zero-padded past Kreal.
__global__ void pack_weights_kernel(const float* __restrict__ W,
                                    _Float16* __restrict__ dst,
                                    int Kreal, int Ncols, int KT, int NT) {
  const int t = blockIdx.x * blockDim.x + threadIdx.x;
  const int total = KT * NT * 32;
  if (t >= total) return;
  const int lane = t & 31;
  const int f = t >> 5;
  const int kt = f / NT, nt = f % NT;
  const int col = nt * 16 + (lane & 15);
  const int kbase = kt * 32 + ((lane >> 4) << 3);
#pragma unroll
  for (int i = 0; i < 16; ++i) {
    const int kr = kbase + (i & 7) + ((i >> 3) << 4);
    const float v = (kr < Kreal) ? W[(size_t)kr * Ncols + col] : 0.0f;
    dst[(size_t)f * 512 + lane * 16 + i] = (_Float16)v;
  }
}

// ================= kernel 0b: f32 -> f16 convert ============================
__global__ void cvt_f16_kernel(const float* __restrict__ src,
                               _Float16* __restrict__ dst, int n) {
  const int t = blockIdx.x * blockDim.x + threadIdx.x;
  if (t < n) dst[t] = (_Float16)src[t];
}

// ================= kernel 1: pair scoring + top-256 selection ===============
__global__ __launch_bounds__(256)
void select_pairs_kernel(const int* __restrict__ z, const float* __restrict__ pos,
                         const unsigned char* __restrict__ mask,
                         const float* __restrict__ z_emb, const int* __restrict__ aidxp,
                         int* __restrict__ jidx, int* __restrict__ kidx,
                         float* __restrict__ wbuf, float* __restrict__ wsum,
                         _Float16* __restrict__ epair) {
  __shared__ float sx[NNODE], sy[NNODE], sz[NNODE], srab[NNODE], scf[NNODE];
  __shared__ unsigned char sval[NNODE];
  __shared__ unsigned char spj[NPAIR], spk[NPAIR];
  __shared__ unsigned long long keys[2048];
  __shared__ float red[256];
  const int b = blockIdx.x;
  const int t = threadIdx.x;
  const int aidx = aidxp[0];

  if (t < NNODE) {
    const float axp = pos[(b*NNODE + aidx)*3 + 0];
    const float ayp = pos[(b*NNODE + aidx)*3 + 1];
    const float azp = pos[(b*NNODE + aidx)*3 + 2];
    const float dx = pos[(b*NNODE + t)*3 + 0] - axp;
    const float dy = pos[(b*NNODE + t)*3 + 1] - ayp;
    const float dz = pos[(b*NNODE + t)*3 + 2] - azp;
    const float r = sqrtf(dx*dx + dy*dy + dz*dz);
    sx[t] = dx; sy[t] = dy; sz[t] = dz;
    srab[t] = r;
    scf[t] = cutoff_fn(r);
    sval[t] = (mask[b*NNODE + t] != 0) && (t != aidx) && (r <= CUTF);
    // triangular index tables (row-major triu order == jnp.triu_indices)
    int off = t*63 - (t*(t-1))/2;
    for (int k = t + 1; k < NNODE; ++k) { spj[off] = (unsigned char)t; spk[off] = (unsigned char)k; ++off; }
  }
  __syncthreads();

  for (int p = t; p < 2048; p += 256) {
    unsigned long long key;
    if (p < NPAIR) {
      const int j = spj[p], k = spk[p];
      float sc;
      if (sval[j] && sval[k]) {
        const float dx = sx[k]-sx[j], dy = sy[k]-sy[j], dz = sz[k]-sz[j];
        sc = srab[j] + srab[k] + 0.5f * sqrtf(dx*dx + dy*dy + dz*dz);
      } else {
        sc = __builtin_inff();
      }
      // positive-float bit pattern is order-preserving; low bits = pair index
      // gives stable tie-break matching jnp.argsort(kind='stable').
      key = ((unsigned long long)__float_as_uint(sc) << 32) | (unsigned int)p;
    } else {
      key = ~0ull;
    }
    keys[p] = key;
  }

  // bitonic sort ascending, 2048 keys
  for (int k = 2; k <= 2048; k <<= 1) {
    for (int j2 = k >> 1; j2 > 0; j2 >>= 1) {
      __syncthreads();
      for (int i = t; i < 2048; i += 256) {
        const int ixj = i ^ j2;
        if (ixj > i) {
          unsigned long long a = keys[i], c = keys[ixj];
          const bool up = ((i & k) == 0);
          if ((a > c) == up) { keys[i] = c; keys[ixj] = a; }
        }
      }
    }
  }
  __syncthreads();

  // emit top-256
  float w = 0.0f;
  {
    const unsigned long long key = keys[t];
    const unsigned int pidx = (unsigned int)(key & 0xffffffffu);
    int j = 0, k = 0;
    if (pidx < (unsigned)NPAIR) {
      j = spj[pidx]; k = spk[pidx];
      if (sval[j] && sval[k]) {
        const float dx = sx[k]-sx[j], dy = sy[k]-sy[j], dz = sz[k]-sz[j];
        const float rjk = sqrtf(dx*dx + dy*dy + dz*dz);
        w = scf[j] * scf[k] * cutoff_fn(rjk);
      }
    } else {
      j = 0; k = 0;   // padding slot, weight 0
    }
    const int gp = b*PMAX_ + t;
    jidx[gp] = j; kidx[gp] = k; wbuf[gp] = w;
    const int zj = z[b*NNODE + j], zk = z[b*NNODE + k];
#pragma unroll
    for (int i = 0; i < 16; ++i) {
      epair[(size_t)gp*32 + i]      = (_Float16)z_emb[zj*16 + i];
      epair[(size_t)gp*32 + 16 + i] = (_Float16)z_emb[zk*16 + i];
    }
  }
  red[t] = w;
  __syncthreads();
  for (int s = 128; s > 0; s >>= 1) { if (t < s) red[t] += red[t + s]; __syncthreads(); }
  if (t == 0) wsum[b] = red[0];
}

// ================= kernel 2: g_geom MLP (353->128->128->32) =================
// One wave per 16-pair M-tile; A padded to K=384; gg = w * (h2@W3 + b3).
// Each B fragment is used exactly once per wave -> stream from L2.
__global__ __launch_bounds__(64)
void geom_kernel(const float* __restrict__ hfeat, const float* __restrict__ pos,
                 const int* __restrict__ jidx, const int* __restrict__ kidx,
                 const float* __restrict__ wbuf,
                 const _Float16* __restrict__ gm1p, const _Float16* __restrict__ gm2p,
                 const _Float16* __restrict__ gm3p,
                 const float* __restrict__ gm_b1, const float* __restrict__ gm_b2,
                 const float* __restrict__ gm_b3,
                 const int* __restrict__ aidxp, float* __restrict__ gg) {
  __shared__ _Float16 smem[2 * 8192];
  const int wid = threadIdx.x >> 5;
  const int lane = threadIdx.x & 31;
  _Float16* A = smem + wid * 8192;   // 16 x 384 features
  _Float16* H = A + 16 * 384;        // 16 x 128 hidden
  const int tile = blockIdx.x * 2 + wid;
  const int b = tile >> 4;
  const int p0 = (tile & 15) << 4;
  const int aidx = aidxp[0];
  const float ax = pos[(b*NNODE + aidx)*3 + 0];
  const float ay = pos[(b*NNODE + aidx)*3 + 1];
  const float az = pos[(b*NNODE + aidx)*3 + 2];
  const float step = CUTF / 31.0f;
  const float coef = -0.5f / (step * step);
  const float roff = step * (float)lane;

  for (int r = 0; r < 16; ++r) {
    const int gp = b*PMAX_ + p0 + r;
    const int j = jidx[gp], k = kidx[gp];
    const float jx = pos[(b*NNODE + j)*3+0]-ax, jy = pos[(b*NNODE + j)*3+1]-ay, jz = pos[(b*NNODE + j)*3+2]-az;
    const float kx = pos[(b*NNODE + k)*3+0]-ax, ky = pos[(b*NNODE + k)*3+1]-ay, kz = pos[(b*NNODE + k)*3+2]-az;
    const float r0j = sqrtf(jx*jx + jy*jy + jz*jz);
    const float r0k = sqrtf(kx*kx + ky*ky + kz*kz);
    const float dx = kx-jx, dy = ky-jy, dz = kz-jz;
    const float rjk = sqrtf(dx*dx + dy*dy + dz*dz);
    const float inj = 1.0f / fmaxf(r0j, 1e-8f);
    const float ink = 1.0f / fmaxf(r0k, 1e-8f);
    const float ca = fminf(1.0f, fmaxf(-1.0f, (jx*kx + jy*ky + jz*kz) * inj * ink));
    const float* hj = hfeat + (size_t)(b*NNODE + j) * DDIM;
    const float* hk = hfeat + (size_t)(b*NNODE + k) * DDIM;
    _Float16* Ar = A + r * 384;
#pragma unroll
    for (int c = 0; c < 4; ++c) {
      Ar[c*32 + lane]       = (_Float16)hj[c*32 + lane];
      Ar[128 + c*32 + lane] = (_Float16)hk[c*32 + lane];
    }
    float t1 = fminf(r0j, CUTF) - roff; Ar[256 + lane] = (_Float16)__expf(coef * t1 * t1);
    float t2 = fminf(r0k, CUTF) - roff; Ar[288 + lane] = (_Float16)__expf(coef * t2 * t2);
    float t3 = fminf(rjk, CUTF) - roff; Ar[320 + lane] = (_Float16)__expf(coef * t3 * t3);
    Ar[352 + lane] = (lane == 0) ? (_Float16)ca : (_Float16)0.0f;  // cos + zero pad
  }
  __syncthreads();

  // layer 1: [16x384] @ [384x128]
  v8f acc[8];
#pragma unroll
  for (int nt = 0; nt < 8; ++nt) acc[nt] = vzero();
  for (int kt = 0; kt < 12; ++kt) {
    const v16h a = load_a_frag(A, lane, 384, kt);
#pragma unroll
    for (int nt = 0; nt < 8; ++nt)
      acc[nt] = wmma32(a, load_b_frag(gm1p, kt*8 + nt, lane), acc[nt]);
  }
  __syncthreads();
  bias_silu_store<8>(acc, H, 128, gm_b1, lane);
  __syncthreads();

  // layer 2: [16x128] @ [128x128]  (output tile stored into A region)
#pragma unroll
  for (int nt = 0; nt < 8; ++nt) acc[nt] = vzero();
  for (int kt = 0; kt < 4; ++kt) {
    const v16h a = load_a_frag(H, lane, 128, kt);
#pragma unroll
    for (int nt = 0; nt < 8; ++nt)
      acc[nt] = wmma32(a, load_b_frag(gm2p, kt*8 + nt, lane), acc[nt]);
  }
  __syncthreads();
  bias_silu_store<8>(acc, A, 384, gm_b2, lane);
  __syncthreads();

  // layer 3: [16x128] @ [128x32] -> gg = w * (x + b3)
  v8f acc3[2];
  acc3[0] = vzero(); acc3[1] = vzero();
  for (int kt = 0; kt < 4; ++kt) {
    const v16h a = load_a_frag(A, lane, 384, kt);
#pragma unroll
    for (int nt = 0; nt < 2; ++nt)
      acc3[nt] = wmma32(a, load_b_frag(gm3p, kt*2 + nt, lane), acc3[nt]);
  }
  const int cl = lane & 15, rb = (lane >> 4) << 3;
#pragma unroll
  for (int nt = 0; nt < 2; ++nt) {
    const int col = nt*16 + cl;
    const float bv = gm_b3[col];
#pragma unroll
    for (int r = 0; r < 8; ++r) {
      const int gp = b*PMAX_ + p0 + rb + r;
      gg[(size_t)gp * SCAT_ + col] = (acc3[nt][r] + bv) * wbuf[gp];
    }
  }
}

// ========== kernel 3: g_elem MLP (48->64->64->32) + pair aggregation ========
// Grid: (batch, NE-chunk of 20). 4 waves split the chunk (5 elements each) and
// loop all 16 pair-chunks. All 20 weight B-fragments are register-resident
// (~160 VGPRs): the inner loop is pure ds_load_b128 + v_wmma.
__global__ __launch_bounds__(128)
void elem_agg_kernel(const _Float16* __restrict__ epair, const _Float16* __restrict__ ef16,
                     const _Float16* __restrict__ pe1p, const _Float16* __restrict__ pe2p,
                     const _Float16* __restrict__ pe3p,
                     const float* __restrict__ pe_b1, const float* __restrict__ pe_b2,
                     const float* __restrict__ pe_b3,
                     const float* __restrict__ gg, float* __restrict__ agg) {
  __shared__ _Float16 smem[4 * 2048];
  __shared__ float aggL[ECHUNK * SCAT_];
  const int b  = blockIdx.x / (NE_ / ECHUNK);
  const int e0 = (blockIdx.x % (NE_ / ECHUNK)) * ECHUNK;
  const int wid = threadIdx.x >> 5;
  const int lane = threadIdx.x & 31;
  _Float16* A = smem + wid * 2048;   // 16 x 64 features
  _Float16* H = A + 1024;            // 16 x 64 hidden

  // weight-stationary B fragments (layer1: 2x4, layer2: 2x4, layer3: 2x2)
  v16h B1[8], B2[8], B3[4];
#pragma unroll
  for (int f = 0; f < 8; ++f) B1[f] = load_b_frag(pe1p, f, lane);
#pragma unroll
  for (int f = 0; f < 8; ++f) B2[f] = load_b_frag(pe2p, f, lane);
#pragma unroll
  for (int f = 0; f < 4; ++f) B3[f] = load_b_frag(pe3p, f, lane);

  for (int i = threadIdx.x; i < ECHUNK * SCAT_; i += 128) aggL[i] = 0.0f;
  __syncthreads();
  const int cl = lane & 15, rb = (lane >> 4) << 3;

  for (int pc = 0; pc < 16; ++pc) {
    const int p0 = pc * 16;
    __builtin_prefetch(&gg[(size_t)(b*PMAX_ + p0) * SCAT_], 0, 0);
    // epair part (cols 0..31) fixed per chunk; cols 32..63 zeroed (48..63 stay 0)
#pragma unroll
    for (int r = 0; r < 16; ++r) {
      A[r*64 + lane]      = epair[(size_t)(b*PMAX_ + p0 + r) * 32 + lane];
      A[r*64 + 32 + lane] = (_Float16)0.0f;
    }
    __syncthreads();

    for (int ei = 0; ei < 5; ++ei) {
      const int el = wid * 5 + ei;       // 0..19 within chunk
      const int e  = e0 + el;
      if (lane < 16) {
        const _Float16 v = ef16[e*16 + lane];
#pragma unroll
        for (int r = 0; r < 16; ++r) A[r*64 + 32 + lane] = v;
      }
      __syncthreads();

      // layer 1: [16x64] @ [64x64]
      v8f acc[4];
#pragma unroll
      for (int nt = 0; nt < 4; ++nt) acc[nt] = vzero();
#pragma unroll
      for (int kt = 0; kt < 2; ++kt) {
        const v16h a = load_a_frag(A, lane, 64, kt);
#pragma unroll
        for (int nt = 0; nt < 4; ++nt)
          acc[nt] = wmma32(a, B1[kt*4 + nt], acc[nt]);
      }
      __syncthreads();
      bias_silu_store<4>(acc, H, 64, pe_b1, lane);
      __syncthreads();

      // layer 2: [16x64] @ [64x64] (in place on H, barrier-guarded)
#pragma unroll
      for (int nt = 0; nt < 4; ++nt) acc[nt] = vzero();
#pragma unroll
      for (int kt = 0; kt < 2; ++kt) {
        const v16h a = load_a_frag(H, lane, 64, kt);
#pragma unroll
        for (int nt = 0; nt < 4; ++nt)
          acc[nt] = wmma32(a, B2[kt*4 + nt], acc[nt]);
      }
      __syncthreads();
      bias_silu_store<4>(acc, H, 64, pe_b2, lane);
      __syncthreads();

      // layer 3: [16x64] @ [64x32]
      v8f acc3[2];
      acc3[0] = vzero(); acc3[1] = vzero();
#pragma unroll
      for (int kt = 0; kt < 2; ++kt) {
        const v16h a = load_a_frag(H, lane, 64, kt);
#pragma unroll
        for (int nt = 0; nt < 2; ++nt)
          acc3[nt] = wmma32(a, B3[kt*2 + nt], acc3[nt]);
      }

      // weighted contribution: sum over 16 pair rows of (g_elem+b3)*gg
#pragma unroll
      for (int nt = 0; nt < 2; ++nt) {
        const int col = nt*16 + cl;
        const float bv = pe_b3[col];
        float part = 0.0f;
#pragma unroll
        for (int r = 0; r < 8; ++r) {
          const int gp = b*PMAX_ + p0 + rb + r;
          part += (acc3[nt][r] + bv) * gg[(size_t)gp * SCAT_ + col];
        }
        part += __shfl_xor(part, 16, 32);   // combine both row halves
        if (lane < 16) aggL[el * SCAT_ + col] += part;
      }
      __syncthreads();
    }
  }
  __syncthreads();
  for (int i = threadIdx.x; i < ECHUNK * SCAT_; i += 128)
    agg[(size_t)b * NE_ * SCAT_ + (size_t)e0 * SCAT_ + i] = aggL[i];
}

// ================= kernel 4: normalize + output MLP (32->128->64) ===========
__global__ __launch_bounds__(128)
void out_mlp_kernel(const float* __restrict__ agg, const float* __restrict__ wsum,
                    const float* __restrict__ W1, const float* __restrict__ b1,
                    const float* __restrict__ W2, const float* __restrict__ b2,
                    float* __restrict__ out) {
  __shared__ float xs[SCAT_];
  __shared__ float hb[GH_];
  const int row = blockIdx.x;          // row = b*NE + e
  const int b = row / NE_;
  const int t = threadIdx.x;
  if (t < SCAT_) xs[t] = agg[(size_t)row * SCAT_ + t] / fmaxf(wsum[b], 1e-8f);
  __syncthreads();
  float a = b1[t];
#pragma unroll 8
  for (int s = 0; s < SCAT_; ++s) a += xs[s] * W1[s * GH_ + t];
  hb[t] = siluf(a);
  __syncthreads();
  if (t < OUTD_) {
    float o = b2[t];
#pragma unroll 8
    for (int k = 0; k < GH_; ++k) o += hb[k] * W2[k * OUTD_ + t];
    out[(size_t)row * OUTD_ + t] = o;
  }
}

// ============================================================================
extern "C" void kernel_launch(void* const* d_in, const int* in_sizes, int n_in,
                              void* d_out, int out_size, void* d_ws, size_t ws_size,
                              hipStream_t stream) {
  const float* hfeat = (const float*)d_in[0];
  const int*   z     = (const int*)d_in[1];
  const float* pos   = (const float*)d_in[2];
  const unsigned char* mask = (const unsigned char*)d_in[3];
  const float* e_feat = (const float*)d_in[4];
  const float* z_emb  = (const float*)d_in[5];
  const float* pe_W1 = (const float*)d_in[6];
  const float* pe_b1 = (const float*)d_in[7];
  const float* pe_W2 = (const float*)d_in[8];
  const float* pe_b2 = (const float*)d_in[9];
  const float* pe_W3 = (const float*)d_in[10];
  const float* pe_b3 = (const float*)d_in[11];
  const float* gm_W1 = (const float*)d_in[12];
  const float* gm_b1 = (const float*)d_in[13];
  const float* gm_W2 = (const float*)d_in[14];
  const float* gm_b2 = (const float*)d_in[15];
  const float* gm_W3 = (const float*)d_in[16];
  const float* gm_b3 = (const float*)d_in[17];
  const float* op_W1 = (const float*)d_in[18];
  const float* op_b1 = (const float*)d_in[19];
  const float* op_W2 = (const float*)d_in[20];
  const float* op_b2 = (const float*)d_in[21];
  const int*   aidx  = (const int*)d_in[22];

  char* ws = (char*)d_ws;
  _Float16* hws = (_Float16*)ws;
  _Float16* pe1p   = hws + H_PE1;
  _Float16* pe2p   = hws + H_PE2;
  _Float16* pe3p   = hws + H_PE3;
  _Float16* gm1p   = hws + H_GM1;
  _Float16* gm2p   = hws + H_GM2;
  _Float16* gm3p   = hws + H_GM3;
  _Float16* efp    = hws + H_EF;
  _Float16* epairp = hws + H_EPAIR;
  float* fws  = (float*)(ws + FBASE);
  float* wbuf = fws + F_W;
  float* wsum = fws + F_WSUM;
  float* gg   = fws + F_GG;
  float* agg  = fws + F_AGG;
  int* jidx = (int*)(ws + IBASE);
  int* kidx = jidx + BDIM * PMAX_;

  // --- pack MLP weights into f16 WMMA B fragments (zero K-padded) ---
  auto pack = [&](const float* W, _Float16* dst, int Kreal, int Ncols, int KT, int NT) {
    const int total = KT * NT * 32;
    pack_weights_kernel<<<(total + 127) / 128, 128, 0, stream>>>(W, dst, Kreal, Ncols, KT, NT);
  };
  pack(pe_W1, pe1p,  48,  64,  2, 4);   // 48 -> pad 64
  pack(pe_W2, pe2p,  64,  64,  2, 4);
  pack(pe_W3, pe3p,  64,  32,  2, 2);
  pack(gm_W1, gm1p, 353, 128, 12, 8);   // 353 -> pad 384
  pack(gm_W2, gm2p, 128, 128,  4, 8);
  pack(gm_W3, gm3p, 128,  32,  4, 2);
  cvt_f16_kernel<<<(NE_*16 + 127) / 128, 128, 0, stream>>>(e_feat, efp, NE_*16);

  // --- pipeline ---
  select_pairs_kernel<<<BDIM, 256, 0, stream>>>(z, pos, mask, z_emb, aidx,
                                                jidx, kidx, wbuf, wsum, epairp);
  geom_kernel<<<(BDIM * PMAX_ / 16) / 2, 64, 0, stream>>>(
      hfeat, pos, jidx, kidx, wbuf, gm1p, gm2p, gm3p, gm_b1, gm_b2, gm_b3, aidx, gg);
  elem_agg_kernel<<<BDIM * (NE_ / ECHUNK), 128, 0, stream>>>(
      epairp, efp, pe1p, pe2p, pe3p, pe_b1, pe_b2, pe_b3, gg, agg);
  out_mlp_kernel<<<BDIM * NE_, 128, 0, stream>>>(agg, wsum, op_W1, op_b1, op_W2, op_b2,
                                                 (float*)d_out);
}
